// MultiheadAttention_51410758533530
// MI455X (gfx1250) — matrix operand order repaired and checked by reading, compiled, not run on hardware
//
#include <hip/hip_runtime.h>
#include <hip/hip_bf16.h>

typedef __attribute__((ext_vector_type(16))) _Float16 v16h;
typedef __attribute__((ext_vector_type(8)))  _Float16 v8h;
typedef __attribute__((ext_vector_type(4)))  _Float16 v4h;
typedef __attribute__((ext_vector_type(8)))  float    v8f;

#define B_  4
#define S_  2048
#define E_  768
#define H_  12
#define D_  64
#define N3_ (3*E_)
#define M_  (B_*S_)

__device__ __forceinline__ v16h cat8(v8h lo, v8h hi) {
    v16h r;
#pragma unroll
    for (int j = 0; j < 8; ++j) { r[j] = lo[j]; r[j + 8] = hi[j]; }
    return r;
}

// Async global->LDS copy, 16B per lane, GVS addressing (saddr64 + vaddr32).
#define ASYNC_LD_B128(ldsoff, voff, base)                                  \
    asm volatile("global_load_async_to_lds_b128 %0, %1, %2"                \
                 :: "v"(ldsoff), "v"(voff),                                \
                    "s"((unsigned long long)(uintptr_t)(base)) : "memory")

// ---------------------------------------------------------------------------
// Kernel 1: qkv = x @ W + b  (f32 in, f16 WMMA, f32 accum)
// Writes Q,K as [B,H,S,D] (f16) and V transposed as [B,H,D,S] (f16).
// Block: 256 thr (8 waves), tile 64(M) x 64(N), K chunk = 32 staged in LDS.
// ---------------------------------------------------------------------------
__global__ __launch_bounds__(256) void qkv_gemm(
    const float* __restrict__ x, const float* __restrict__ W,
    const float* __restrict__ bias,
    _Float16* __restrict__ Qh, _Float16* __restrict__ Kh,
    _Float16* __restrict__ Vt)
{
    __shared__ _Float16 As[64][40];   // [m][k]  (pad 8 halves)
    __shared__ _Float16 Bs[64][40];   // [n][k]  transposed W tile

    const int tid  = threadIdx.x;
    const int lane = tid & 31;
    const int wave = tid >> 5;
    const int hs   = lane >> 4;       // half-wave select
    const int ln   = lane & 15;

    const int blockM = blockIdx.x * 64;
    const int blockN = blockIdx.y * 64;

    const int mi    = wave & 3;       // wave's 16-row tile
    const int nbase = (wave >> 2) * 32;

    v8f acc0 = {}; v8f acc1 = {};

    for (int kb = 0; kb < E_; kb += 32) {
        // ---- stage A tile: 64x32 f32 -> f16 ----
#pragma unroll
        for (int i = 0; i < 2; ++i) {
            int idx = tid + i * 256;                 // 0..511 float4's
            int row = idx >> 3;
            int c4  = (idx & 7) * 4;
            float4 v = *(const float4*)(x + (size_t)(blockM + row) * E_ + kb + c4);
            v4h hv;
            hv[0] = (_Float16)v.x; hv[1] = (_Float16)v.y;
            hv[2] = (_Float16)v.z; hv[3] = (_Float16)v.w;
            *(v4h*)&As[row][c4] = hv;
        }
        // ---- stage B tile transposed: W[k][n] -> Bs[n][k] ----
#pragma unroll
        for (int i = 0; i < 2; ++i) {
            int idx  = tid + i * 256;
            int krow = idx >> 4;                     // 0..31
            int c4   = (idx & 15) * 4;               // 0..60
            float4 v = *(const float4*)(W + (size_t)(kb + krow) * N3_ + blockN + c4);
            Bs[c4 + 0][krow] = (_Float16)v.x;
            Bs[c4 + 1][krow] = (_Float16)v.y;
            Bs[c4 + 2][krow] = (_Float16)v.z;
            Bs[c4 + 3][krow] = (_Float16)v.w;
        }
        __syncthreads();

        // A fragment (16x32, lane=row, K = 16*hs..16*hs+15)
        const int m = mi * 16 + ln;
        v16h af = cat8(*(const v8h*)&As[m][16 * hs],
                       *(const v8h*)&As[m][16 * hs + 8]);

        // B fragments (32x16, lane=col, K = 16*hs..16*hs+15, same k order as A)
        const int n0 = nbase + ln;
        v16h bf0 = cat8(*(const v8h*)&Bs[n0     ][16 * hs],
                        *(const v8h*)&Bs[n0     ][16 * hs + 8]);
        v16h bf1 = cat8(*(const v8h*)&Bs[n0 + 16][16 * hs],
                        *(const v8h*)&Bs[n0 + 16][16 * hs + 8]);

        acc0 = __builtin_amdgcn_wmma_f32_16x16x32_f16(false, af, false, bf0,
                                                      (short)0, acc0, false, false);
        acc1 = __builtin_amdgcn_wmma_f32_16x16x32_f16(false, af, false, bf1,
                                                      (short)0, acc1, false, false);
        __syncthreads();
    }

    // ---- epilogue: bias, scatter to Q / K / V^T ----
    const int mbase = blockM + mi * 16 + 8 * hs;     // C layout: reg r -> m = r + 8*hs
#pragma unroll
    for (int tile = 0; tile < 2; ++tile) {
        int gn   = blockN + nbase + tile * 16 + ln;  // global column in [0,3E)
        float bv = bias[gn];
        int g    = gn / E_;                          // 0=Q 1=K 2=V (uniform per tile)
        int rem  = gn - g * E_;
        int hh   = rem >> 6;
        int d    = rem & 63;
        v8f acc  = tile ? acc1 : acc0;
#pragma unroll
        for (int r = 0; r < 8; ++r) {
            int gm = mbase + r;
            int bb = gm >> 11;                       // / S_
            int s  = gm & (S_ - 1);
            float val = acc[r] + bv;
            size_t head = (size_t)(bb * H_ + hh);
            if (g == 0)      Qh[(head * S_ + s) * D_ + d] = (_Float16)val;
            else if (g == 1) Kh[(head * S_ + s) * D_ + d] = (_Float16)val;
            else             Vt[(head * D_ + d) * S_ + s] = (_Float16)val;
        }
    }
}

// ---------------------------------------------------------------------------
// Kernel 2: flash attention. Block = 128 thr (4 waves), wave owns 16 rows.
// grid = (B*H, S/64). K/V chunks stream into LDS via async B128 copies,
// double-buffered behind the WMMAs (ASYNCcnt), shared by all 4 waves.
// ---------------------------------------------------------------------------
__global__ __launch_bounds__(128) void attn(
    const _Float16* __restrict__ Qh, const _Float16* __restrict__ Kh,
    const _Float16* __restrict__ Vt, float* __restrict__ out)
{
    __shared__ __align__(16) _Float16 Klds[2][32][64];   // [buf][t][d]  4KB each
    __shared__ __align__(16) _Float16 Vlds[2][64][32];   // [buf][d][t]  4KB each
    __shared__ __align__(16) _Float16 Plds[4][16 * 32];  // wave-private P tiles

    const int tid  = threadIdx.x;
    const int lane = tid & 31;
    const int wave = tid >> 5;
    const int hs   = lane >> 4;
    const int ln   = lane & 15;

    const int bh    = blockIdx.x;                    // b*H + h
    const int sBase = blockIdx.y * 64 + wave * 16;

    const _Float16* Qp = Qh + (size_t)bh * S_ * D_;
    const _Float16* Kp = Kh + (size_t)bh * S_ * D_;
    const _Float16* Vp = Vt + (size_t)bh * D_ * S_;  // [d][s]

    const unsigned kbase = (unsigned)(uintptr_t)&Klds[0][0][0];
    const unsigned vbase = (unsigned)(uintptr_t)&Vlds[0][0][0];

    // Issue one chunk's async copies: 4 x B128 per thread (2 K rows-worth, 2 V).
    // K chunk: contiguous 4KB at Kp + t*D.  V chunk: 64 rows x 64B, row stride S*2.
    auto issue = [&](int buf, int t) {
        const _Float16* kt = Kp + (size_t)t * D_;
        const _Float16* vt = Vp + t;
        unsigned kdst = kbase + (unsigned)buf * 4096;
        unsigned vdst = vbase + (unsigned)buf * 4096;
#pragma unroll
        for (int j = 0; j < 2; ++j) {
            unsigned idx  = (unsigned)(tid + j * 128);     // 0..255
            unsigned koff = idx * 16;
            ASYNC_LD_B128(kdst + koff, koff, kt);
            unsigned d    = idx >> 2, part = idx & 3;
            unsigned voff = d * (S_ * 2) + part * 16;
            ASYNC_LD_B128(vdst + d * 64 + part * 16, voff, vt);
        }
    };

    // Q fragments (A layout), d 0..31 and 32..63
    const _Float16* qrow = Qp + (size_t)(sBase + ln) * D_;
    v16h qf0 = cat8(*(const v8h*)(qrow +      16 * hs),
                    *(const v8h*)(qrow +  8 + 16 * hs));
    v16h qf1 = cat8(*(const v8h*)(qrow + 32 + 16 * hs),
                    *(const v8h*)(qrow + 40 + 16 * hs));

    float mrow[8], lrow[8];
    v8f o[4] = {v8f{}, v8f{}, v8f{}, v8f{}};
#pragma unroll
    for (int r = 0; r < 8; ++r) { mrow[r] = -1e30f; lrow[r] = 0.0f; }

    _Float16* Pw = &Plds[wave][0];
    const float scale = 0.125f;                      // 1/sqrt(64)

    issue(0, 0);                                     // prologue: chunk 0

    const int nT = S_ / 32;                          // 64 chunks
    for (int tb = 0; tb < nT; ++tb) {
        const int cur = tb & 1;
        if (tb < nT - 1) {
            issue(cur ^ 1, (tb + 1) * 32);           // stream next chunk behind compute
            // async loads complete in order: <=4 outstanding => current buffer done
            asm volatile("s_wait_asynccnt 4" ::: "memory");
        } else {
            asm volatile("s_wait_asynccnt 0" ::: "memory");
        }
        __syncthreads();

        // ---- scores for two 16-wide t tiles (K frags from LDS) ----
        v8f c0 = {}; v8f c1 = {};
        {
            const _Float16* kr = &Klds[cur][ln][0];
            v16h kfa = cat8(*(const v8h*)(kr +      16 * hs),
                            *(const v8h*)(kr +  8 + 16 * hs));
            v16h kfb = cat8(*(const v8h*)(kr + 32 + 16 * hs),
                            *(const v8h*)(kr + 40 + 16 * hs));
            c0 = __builtin_amdgcn_wmma_f32_16x16x32_f16(false, qf0, false, kfa,
                                                        (short)0, c0, false, false);
            c0 = __builtin_amdgcn_wmma_f32_16x16x32_f16(false, qf1, false, kfb,
                                                        (short)0, c0, false, false);
        }
        {
            const _Float16* kr = &Klds[cur][16 + ln][0];
            v16h kfa = cat8(*(const v8h*)(kr +      16 * hs),
                            *(const v8h*)(kr +  8 + 16 * hs));
            v16h kfb = cat8(*(const v8h*)(kr + 32 + 16 * hs),
                            *(const v8h*)(kr + 40 + 16 * hs));
            c1 = __builtin_amdgcn_wmma_f32_16x16x32_f16(false, qf0, false, kfa,
                                                        (short)0, c1, false, false);
            c1 = __builtin_amdgcn_wmma_f32_16x16x32_f16(false, qf1, false, kfb,
                                                        (short)0, c1, false, false);
        }

        // ---- online softmax; rows live in reg r (m = r + 8*hs), cols across 16 lanes
#pragma unroll
        for (int r = 0; r < 8; ++r) {
            float s0 = c0[r] * scale, s1 = c1[r] * scale;
            float cm = fmaxf(s0, s1);
#pragma unroll
            for (int msk = 1; msk < 16; msk <<= 1)
                cm = fmaxf(cm, __shfl_xor(cm, msk, 32));
            float nm   = fmaxf(mrow[r], cm);
            float corr = __expf(mrow[r] - nm);
            mrow[r] = nm;
            float p0 = __expf(s0 - nm), p1 = __expf(s1 - nm);
            float ps = p0 + p1;
#pragma unroll
            for (int msk = 1; msk < 16; msk <<= 1)
                ps += __shfl_xor(ps, msk, 32);
            lrow[r] = lrow[r] * corr + ps;
            o[0][r] *= corr; o[1][r] *= corr; o[2][r] *= corr; o[3][r] *= corr;
            int m = r + 8 * hs;                      // C-layout row -> P row
            Pw[m * 32 + ln]      = (_Float16)p0;
            Pw[m * 32 + 16 + ln] = (_Float16)p1;
        }

        // in-wave LDS RAW: wait for P stores, block compiler reordering
        asm volatile("s_wait_dscnt 0" ::: "memory");

        // reload P in A layout (lane = row ln, K = 16*hs..16*hs+15)
        v16h pf = cat8(*(const v8h*)&Pw[ln * 32 +     16 * hs],
                       *(const v8h*)&Pw[ln * 32 + 8 + 16 * hs]);

        // ---- O += P @ V : V^T rows from LDS, contiguous in t ----
#pragma unroll
        for (int i = 0; i < 4; ++i) {
            const _Float16* vr = &Vlds[cur][i * 16 + ln][16 * hs];
            v16h vf = cat8(*(const v8h*)vr, *(const v8h*)(vr + 8));
            o[i] = __builtin_amdgcn_wmma_f32_16x16x32_f16(false, pf, false, vf,
                                                          (short)0, o[i], false, false);
        }

        __syncthreads();   // all waves done with 'cur' before it is refilled
    }

    // ---- normalize and store [B,S,E] f32 ----
    const int bg = bh / H_;
    const int hh = bh % H_;
    float* obase = out + (size_t)bg * S_ * E_ + hh * D_;
#pragma unroll
    for (int r = 0; r < 8; ++r) {
        float inv = 1.0f / lrow[r];
        int s = sBase + r + 8 * hs;
        float* orow = obase + (size_t)s * E_;
        orow[ 0 + ln] = o[0][r] * inv;
        orow[16 + ln] = o[1][r] * inv;
        orow[32 + ln] = o[2][r] * inv;
        orow[48 + ln] = o[3][r] * inv;
    }
}

// ---------------------------------------------------------------------------
extern "C" void kernel_launch(void* const* d_in, const int* in_sizes, int n_in,
                              void* d_out, int out_size, void* d_ws, size_t ws_size,
                              hipStream_t stream) {
    const float* x    = (const float*)d_in[0];
    const float* W    = (const float*)d_in[1];
    const float* bias = (const float*)d_in[2];
    float* out        = (float*)d_out;

    const size_t per = (size_t)B_ * H_ * S_ * D_;    // elements per tensor
    _Float16* Qh = (_Float16*)d_ws;
    _Float16* Kh = Qh + per;
    _Float16* Vt = Kh + per;

    dim3 g1(M_ / 64, N3_ / 64), b1(256);
    qkv_gemm<<<g1, b1, 0, stream>>>(x, W, bias, Qh, Kh, Vt);

    dim3 g2(B_ * H_, S_ / 64), b2(128);
    attn<<<g2, b2, 0, stream>>>(Qh, Kh, Vt, out);
}
